// NeighborDiscriminator_19447611916838
// MI455X (gfx1250) — compile-verified, matrix operand order checked by reading, not run on hardware
//
#include <hip/hip_runtime.h>

// ---------------- problem constants ----------------
#define N_DB   100000   // database rows
#define DIMK   128      // feature dim
#define M_Q    2048     // queries
#define K_TOP  10       // top-k
#define TPW    6        // N-tiles per wave
#define CHUNK  768      // N-candidates per workgroup (48 tiles of 16)
#define NCHUNK 131      // ceil(100000/768)
#define N_PAD  (NCHUNK * CHUNK)   // 100608 rows (padded so B addressing is pure immediates)
#define PITCH  772      // LDS key-tile pitch (772%64==4 -> conflict-free bank windows)
#define APITCH 136      // A-tile row pitch in elements (272B, b128-aligned, stride-4 banks)

typedef __bf16  bf16_t;
typedef bf16_t  v16bf __attribute__((ext_vector_type(16)));
typedef float   v8f   __attribute__((ext_vector_type(8)));
typedef unsigned int u32x4 __attribute__((ext_vector_type(4)));
typedef unsigned int u32x2 __attribute__((ext_vector_type(2)));
typedef float   f32x4 __attribute__((ext_vector_type(4)));

union FragB { u32x4 q[2]; v16bf v; };

// scheduling-group hints (no-op if builtin unavailable)
#if __has_builtin(__builtin_amdgcn_sched_group_barrier)
#define SGB(mask, n) __builtin_amdgcn_sched_group_barrier((mask), (n), 0)
#else
#define SGB(mask, n)
#endif

// ---------------- workspace layout (bytes) ----------------
static constexpr size_t XHI_OFF   = 0;
static constexpr size_t XLO_OFF   = XHI_OFF   + (size_t)N_PAD * DIMK * 2;  // 25.76 MB
static constexpr size_t AHI_OFF   = XLO_OFF   + (size_t)N_PAD * DIMK * 2;  // 51.51 MB
static constexpr size_t ALO_OFF   = AHI_OFF   + (size_t)M_Q  * DIMK * 2;
static constexpr size_t R_OFF     = ALO_OFF   + (size_t)M_Q  * DIMK * 2;
static constexpr size_t Q2_OFF    = R_OFF     + (size_t)N_DB * 4;
static constexpr size_t WPART_OFF = Q2_OFF    + (size_t)M_Q  * 4;
static constexpr size_t WMAX_OFF  = WPART_OFF + 512;
static constexpr size_t CANDD_OFF = WMAX_OFF  + 64;
static constexpr size_t CANDI_OFF = CANDD_OFF + (size_t)M_Q * NCHUNK * K_TOP * 4;
// total ≈ 74 MB of d_ws required

__device__ __forceinline__ unsigned int bf16_rne(float f) {
  unsigned int u = __float_as_uint(f);
  return (u + 0x7FFFu + ((u >> 16) & 1u)) >> 16;   // round-to-nearest-even
}

__device__ __forceinline__ void loadFrag(FragB& f, const unsigned short* p) {
  f.q[0] = *(const u32x4*)(p);
  f.q[1] = *(const u32x4*)(p + 16);
}

// ---------------- stage 1a: per-block max(relu(w)) ----------------
__global__ void k_wmax_part(const float* __restrict__ w, float* __restrict__ part, int n) {
  __shared__ float s[256];
  int t = threadIdx.x;
  int base = blockIdx.x * 1024 + t;
  float m = 0.0f;                       // relu floor
  #pragma unroll
  for (int i = 0; i < 4; ++i) {
    int idx = base + i * 256;
    if (idx < n) m = fmaxf(m, w[idx]);
  }
  s[t] = m; __syncthreads();
  for (int off = 128; off > 0; off >>= 1) {
    if (t < off) s[t] = fmaxf(s[t], s[t + off]);
    __syncthreads();
  }
  if (t == 0) part[blockIdx.x] = s[0];
}

// ---------------- stage 1b: final wmax ----------------
__global__ void k_wmax_final(const float* __restrict__ part, float* __restrict__ wmax, int nparts) {
  __shared__ float s[128];
  int t = threadIdx.x;
  s[t] = (t < nparts) ? part[t] : 0.0f;
  __syncthreads();
  for (int off = 64; off > 0; off >>= 1) {
    if (t < off) s[t] = fmaxf(s[t], s[t + off]);
    __syncthreads();
  }
  if (t == 0) wmax[0] = s[0];
}

// ---------------- stage 2: fp32 -> split bf16 (hi+lo) + row norms ----------------
// mode 0: aux[row] = ||row||^2 + wmax - w[row]   (database rows)
// mode 1: aux[row] = ||row||^2                   (query rows)
__global__ void k_prep(const float* __restrict__ src, const float* __restrict__ w,
                       const float* __restrict__ wmaxp,
                       unsigned short* __restrict__ hi, unsigned short* __restrict__ lo,
                       float* __restrict__ aux, int nrows, int mode) {
  int row  = blockIdx.x * 8 + (threadIdx.x >> 5);
  int lane = threadIdx.x & 31;
  if (row >= nrows) return;               // wave-uniform
  size_t base = (size_t)row * DIMK + lane * 4;
  f32x4 v = *(const f32x4*)(src + base);

  unsigned int h0 = bf16_rne(v.x), h1 = bf16_rne(v.y), h2 = bf16_rne(v.z), h3 = bf16_rne(v.w);
  unsigned int l0 = bf16_rne(v.x - __uint_as_float(h0 << 16));
  unsigned int l1 = bf16_rne(v.y - __uint_as_float(h1 << 16));
  unsigned int l2 = bf16_rne(v.z - __uint_as_float(h2 << 16));
  unsigned int l3 = bf16_rne(v.w - __uint_as_float(h3 << 16));

  u32x2 ph, pl;
  ph.x = h0 | (h1 << 16); ph.y = h2 | (h3 << 16);
  pl.x = l0 | (l1 << 16); pl.y = l2 | (l3 << 16);
  *(u32x2*)(hi + base) = ph;
  *(u32x2*)(lo + base) = pl;

  float ss = v.x * v.x + v.y * v.y + v.z * v.z + v.w * v.w;
  #pragma unroll
  for (int off = 16; off > 0; off >>= 1) ss += __shfl_xor(ss, off, 32);
  if (lane == 0) {
    if (mode == 0) aux[row] = ss + wmaxp[0] - w[row];   // ||x_n||^2 + w'_n^2
    else           aux[row] = ss;                        // ||q||^2
  }
}

// ---- write one 16-row half of keys into the LDS tile ----
__device__ __forceinline__ void store_keys(float* lds, const v8f* acc,
                                           const float* __restrict__ rv_arr,
                                           int nbase, int wave, int idx, int khalf) {
  #pragma unroll
  for (int t = 0; t < TPW; ++t) {
    int colBase = (wave * TPW + t) * 16 + idx;
    int n = nbase + colBase;
    float rv = (n < N_DB) ? rv_arr[n] : __builtin_inff();  // pad cols -> +inf (NaN-safe too)
    #pragma unroll
    for (int rr = 0; rr < 8; ++rr) {
      lds[(rr + 8 * khalf) * PITCH + colBase] = fmaf(-2.0f, acc[t][rr], rv);
    }
  }
}

// ---- fused two-level top-10 over the 16 x CHUNK LDS key tile, emit to scratch ----
__device__ __forceinline__ void topk_select_emit(float* lds, int nbase, int gqBase,
                                                 int chunkIdx,
                                                 float* __restrict__ candD,
                                                 int* __restrict__ candI) {
  const int tid  = threadIdx.x;
  const int q    = tid & 15;
  const int part = tid >> 4;
  float td[K_TOP]; int ti[K_TOP];
  #pragma unroll
  for (int j = 0; j < K_TOP; ++j) { td[j] = __builtin_inff(); ti[j] = 0; }
  for (int j = 0; j < CHUNK / 16; ++j) {       // interleaved columns: bank-conflict-free
    int col = part + 16 * j;
    float d = lds[q * PITCH + col];
    if (d < td[K_TOP - 1]) {                   // rarely-taken guarded insert
      int id = nbase + col;
      #pragma unroll
      for (int s2 = 0; s2 < K_TOP; ++s2) {
        float od = td[s2]; int oi = ti[s2];
        bool c = d < od;
        td[s2] = c ? d : od; ti[s2] = c ? id : oi;
        d = c ? od : d;      id = c ? oi : id;
      }
    }
  }
  __syncthreads();
  float* listD = lds;
  int*   listI = (int*)(lds + 256 * K_TOP);
  #pragma unroll
  for (int j = 0; j < K_TOP; ++j) { listD[tid * K_TOP + j] = td[j]; listI[tid * K_TOP + j] = ti[j]; }
  __syncthreads();
  if (tid < 16) {                              // merge 16 sorted sublists per query
    float md[K_TOP]; int mi[K_TOP];
    #pragma unroll
    for (int j = 0; j < K_TOP; ++j) { md[j] = __builtin_inff(); mi[j] = 0; }
    for (int p = 0; p < 16; ++p) {
      int src = (p * 16 + tid) * K_TOP;
      for (int j = 0; j < K_TOP; ++j) {
        float d = listD[src + j];
        if (!(d < md[K_TOP - 1])) break;       // sublists sorted ascending
        int id = listI[src + j];
        #pragma unroll
        for (int s2 = 0; s2 < K_TOP; ++s2) {
          float od = md[s2]; int oi = mi[s2];
          bool c = d < od;
          md[s2] = c ? d : od; mi[s2] = c ? id : oi;
          d = c ? od : d;      id = c ? oi : id;
        }
      }
    }
    size_t o = ((size_t)(gqBase + tid) * NCHUNK + chunkIdx) * K_TOP;
    #pragma unroll
    for (int j = 0; j < K_TOP; ++j) { candD[o + j] = md[j]; candI[o + j] = mi[j]; }
  }
}

// ---------------- stage 3: WMMA GEMM (split-bf16) + fused per-chunk top-10 ----------------
// grid: (NCHUNK, M_Q/32). block: 256 (8 waves). Workgroup = 32 queries x 768 candidates.
// A tile staged in LDS; B prefetched in half-k-step groups of 3 tiles, double-buffered.
// __launch_bounds__(256,2): allow ~half the VGPR file per wave so all buffers stay live.
// sched_group_barrier pins the pipeline: [A ds-loads][next B group loads][18 WMMAs].
__global__ void __launch_bounds__(256, 2) k_gemm_topk(
    const unsigned short* __restrict__ Ahi, const unsigned short* __restrict__ Alo,
    const unsigned short* __restrict__ Xhi, const unsigned short* __restrict__ Xlo,
    const float* __restrict__ rv_arr, float* __restrict__ candD, int* __restrict__ candI) {
  __shared__ float lds[16 * PITCH];                 // 49.4 KB: A tile (17.4 KB) aliased, then key tile
  const int tid   = threadIdx.x;
  const int lane  = tid & 31;
  const int wave  = tid >> 5;
  const int idx   = lane & 15;                      // row-of-A / col-of-B
  const int khalf = lane >> 4;                      // K-half select (wave32 frag layout)
  const int mbase = blockIdx.y * 32;
  const int nbase = blockIdx.x * CHUNK;

  // ---- stage A tile (32 rows x 128, hi+lo) into LDS, padded pitch ----
  unsigned short* aHiT = (unsigned short*)lds;            // 32*APITCH elems
  unsigned short* aLoT = aHiT + 32 * APITCH;
  {
    const int r = tid >> 3;          // 0..31
    const int seg = tid & 7;         // 8 segments of 16 elems per 128-elem row
    const unsigned short* gh = Ahi + (size_t)(mbase + r) * DIMK + seg * 16;
    const unsigned short* gl = Alo + (size_t)(mbase + r) * DIMK + seg * 16;
    u32x4 h0 = *(const u32x4*)(gh), h1 = *(const u32x4*)(gh + 8);
    u32x4 l0 = *(const u32x4*)(gl), l1 = *(const u32x4*)(gl + 8);
    unsigned short* dh = aHiT + r * APITCH + seg * 16;
    unsigned short* dl = aLoT + r * APITCH + seg * 16;
    *(u32x4*)(dh) = h0; *(u32x4*)(dh + 8) = h1;
    *(u32x4*)(dl) = l0; *(u32x4*)(dl + 8) = l1;
  }
  __syncthreads();

  v8f acc0[TPW], acc1[TPW];
  #pragma unroll
  for (int t = 0; t < TPW; ++t) {
    #pragma unroll
    for (int i = 0; i < 8; ++i) { acc0[t][i] = 0.0f; acc1[t][i] = 0.0f; }
  }

  const size_t bRow0 = (size_t)(nbase + wave * (TPW * 16) + idx) * DIMK + khalf * 8;
  const unsigned short* bHi0 = Xhi + bRow0;
  const unsigned short* bLo0 = Xlo + bRow0;

  FragB pbh[2][3], pbl[2][3];          // B prefetch: 2 buffers x 3 tiles (96 VGPRs)
  FragB ah0, al0, ah1, al1;            // A frags for current k-step (from LDS)

  // prologue: B half-group h=0 (s=0, tiles 0..2)
  #pragma unroll
  for (int i = 0; i < 3; ++i) {
    loadFrag(pbh[0][i], bHi0 + i * 16 * DIMK);
    loadFrag(pbl[0][i], bLo0 + i * 16 * DIMK);
  }

  #pragma unroll
  for (int h = 0; h < 8; ++h) {                    // 4 k-steps x 2 half-groups
    const int s = h >> 1, half = h & 1, buf = h & 1;
    if (half == 0) {                               // A frags for this k-step from LDS
      const int ao = s * 32 + khalf * 8;
      loadFrag(ah0, aHiT + idx * APITCH + ao);
      loadFrag(al0, aLoT + idx * APITCH + ao);
      loadFrag(ah1, aHiT + (16 + idx) * APITCH + ao);
      loadFrag(al1, aLoT + (16 + idx) * APITCH + ao);
    }
    if (h + 1 < 8) {                               // prefetch next B half-group
      const int h2 = h + 1, s2 = h2 >> 1, half2 = h2 & 1, buf2 = h2 & 1;
      #pragma unroll
      for (int i = 0; i < 3; ++i) {
        const int to = s2 * 32 + (half2 * 3 + i) * 16 * DIMK;   // immediates
        loadFrag(pbh[buf2][i], bHi0 + to);
        loadFrag(pbl[buf2][i], bLo0 + to);
      }
    }
    #pragma unroll
    for (int i = 0; i < 3; ++i) {
      const int t = half * 3 + i;
      // split-bf16: hi*hi + hi*lo + lo*hi per M-half; B frags shared across halves
      acc0[t] = __builtin_amdgcn_wmma_f32_16x16x32_bf16(false, ah0.v, false, pbh[buf][i].v, (short)0, acc0[t], false, false);
      acc0[t] = __builtin_amdgcn_wmma_f32_16x16x32_bf16(false, ah0.v, false, pbl[buf][i].v, (short)0, acc0[t], false, false);
      acc0[t] = __builtin_amdgcn_wmma_f32_16x16x32_bf16(false, al0.v, false, pbh[buf][i].v, (short)0, acc0[t], false, false);
      acc1[t] = __builtin_amdgcn_wmma_f32_16x16x32_bf16(false, ah1.v, false, pbh[buf][i].v, (short)0, acc1[t], false, false);
      acc1[t] = __builtin_amdgcn_wmma_f32_16x16x32_bf16(false, ah1.v, false, pbl[buf][i].v, (short)0, acc1[t], false, false);
      acc1[t] = __builtin_amdgcn_wmma_f32_16x16x32_bf16(false, al1.v, false, pbh[buf][i].v, (short)0, acc1[t], false, false);
    }
    // pin pipeline shape: A ds-loads, then next-group VMEM loads, then this group's WMMAs
    if (half == 0) SGB(0x100, 8);                  // DS reads (A frags)
    if (h + 1 < 8) SGB(0x020, 12);                 // VMEM reads (next B half-group)
    SGB(0x008, 18);                                // WMMA group
  }
  __syncthreads();                                 // A tile dead; lds becomes key tile

  // ---- M-half 0: queries [mbase, mbase+16) ----
  store_keys(lds, acc0, rv_arr, nbase, wave, idx, khalf);
  __syncthreads();
  topk_select_emit(lds, nbase, mbase, blockIdx.x, candD, candI);
  __syncthreads();
  // ---- M-half 1: queries [mbase+16, mbase+32) ----
  store_keys(lds, acc1, rv_arr, nbase, wave, idx, khalf);
  __syncthreads();
  topk_select_emit(lds, nbase, mbase + 16, blockIdx.x, candD, candI);
}

// ---------------- stage 4: merge chunks, rescore k=10, argmax ----------------
__global__ void k_final(const float* __restrict__ candD, const int* __restrict__ candI,
                        const float* __restrict__ q2, const float* __restrict__ w,
                        const float* __restrict__ wmaxp, float* __restrict__ out) {
  int m = blockIdx.x * blockDim.x + threadIdx.x;
  if (m >= M_Q) return;
  const float wmax = wmaxp[0];
  float td[K_TOP]; int ti[K_TOP];
  #pragma unroll
  for (int j = 0; j < K_TOP; ++j) { td[j] = __builtin_inff(); ti[j] = 0; }
  size_t base = (size_t)m * NCHUNK * K_TOP;
  for (int e = 0; e < NCHUNK * K_TOP; ++e) {
    float d = candD[base + e];
    if (d < td[K_TOP - 1]) {
      int id = candI[base + e];
      #pragma unroll
      for (int s2 = 0; s2 < K_TOP; ++s2) {
        float od = td[s2]; int oi = ti[s2];
        bool c = d < od;
        td[s2] = c ? d : od; ti[s2] = c ? id : oi;
        d = c ? od : d;      id = c ? oi : id;
      }
    }
  }
  float q2m = q2[m];
  float bestF = -__builtin_inff(); int bestI = 0;
  #pragma unroll
  for (int j = 0; j < K_TOP; ++j) {
    float D   = td[j] + q2m;                 // full squared augmented distance
    float wn  = w[ti[j]];
    float wp2 = wmax - wn;                   // w'^2 (K=1)
    float da  = sqrtf(fmaxf(D - wp2, 0.0f)); // actual distance
    float f   = wn - da;                     // f_i = w - K*dist
    if (f > bestF) { bestF = f; bestI = ti[j]; }
  }
  out[m]       = bestF;
  out[M_Q + m] = (float)bestI;               // index output (exact in fp32: < 2^24)
}

extern "C" void kernel_launch(void* const* d_in, const int* in_sizes, int n_in,
                              void* d_out, int out_size, void* d_ws, size_t ws_size,
                              hipStream_t stream) {
  const float* x_tilde = (const float*)d_in[0];   // [2048,128]
  const float* X       = (const float*)d_in[1];   // [100000,128]
  const float* w       = (const float*)d_in[2];   // [100000,1]
  // d_in[3] = k (==10), baked in as K_TOP
  (void)in_sizes; (void)n_in; (void)out_size; (void)ws_size;

  char* ws = (char*)d_ws;
  unsigned short* Xhi   = (unsigned short*)(ws + XHI_OFF);
  unsigned short* Xlo   = (unsigned short*)(ws + XLO_OFF);
  unsigned short* Ahi   = (unsigned short*)(ws + AHI_OFF);
  unsigned short* Alo   = (unsigned short*)(ws + ALO_OFF);
  float*          rbuf  = (float*)(ws + R_OFF);
  float*          q2    = (float*)(ws + Q2_OFF);
  float*          wpart = (float*)(ws + WPART_OFF);
  float*          wmaxp = (float*)(ws + WMAX_OFF);
  float*          candD = (float*)(ws + CANDD_OFF);
  int*            candI = (int*)(ws + CANDI_OFF);

  k_wmax_part <<<98, 256, 0, stream>>>(w, wpart, N_DB);
  k_wmax_final<<<1, 128, 0, stream>>>(wpart, wmaxp, 98);
  k_prep      <<<(N_DB + 7) / 8, 256, 0, stream>>>(X, w, wmaxp, Xhi, Xlo, rbuf, N_DB, 0);
  k_prep      <<<M_Q / 8, 256, 0, stream>>>(x_tilde, nullptr, wmaxp, Ahi, Alo, q2, M_Q, 1);
  dim3 g(NCHUNK, M_Q / 32);
  k_gemm_topk <<<g, 256, 0, stream>>>(Ahi, Alo, Xhi, Xlo, rbuf, candD, candI);
  k_final     <<<M_Q / 256, 256, 0, stream>>>(candD, candI, q2, w, wmaxp, (float*)d_out);
}
